// RNTN_33157147525838
// MI455X (gfx1250) — matrix operand order repaired and checked by reading, not compiled, not run on hardware
//
#include <hip/hip_runtime.h>

// RNTN forward for MI455X (gfx1250, wave32).
// Bilinear xVx + linear cW recast as one bf16 WMMA GEMM with K=4096+64,
// A-fragments built on the fly with v_pk_mul_bf16 from LDS-staged children.
// All fragments are ext-vector SSA values (no unions) so they stay in VGPRs.

#define Bt    8192
#define Lt    64
#define Ht    32
#define OUTC  5
#define NSLAB 130   // 128 bilinear K-slabs (K=4096) + 2 linear slabs (W, K=64)

typedef __attribute__((ext_vector_type(16))) __bf16        v16bf;
typedef __attribute__((ext_vector_type(8)))  float         v8f;
typedef __attribute__((ext_vector_type(8)))  unsigned int  v8u;

__device__ __forceinline__ unsigned int f2bf(float x) {           // RNE f32->bf16
  unsigned int u = __float_as_uint(x);
  return (u + 0x7FFFu + ((u >> 16) & 1u)) >> 16;
}
__device__ __forceinline__ unsigned int pk2(float lo, float hi) {
  return f2bf(lo) | (f2bf(hi) << 16);
}
__device__ __forceinline__ unsigned int pkmul_bf16(unsigned int a, unsigned int b) {
  unsigned int d;
  asm("v_pk_mul_bf16 %0, %1, %2" : "=v"(d) : "v"(a), "v"(b));
  return d;
}
__device__ __forceinline__ v8u frag_from(uint4 a, uint4 b) {
  v8u r;
  r[0] = a.x; r[1] = a.y; r[2] = a.z; r[3] = a.w;
  r[4] = b.x; r[5] = b.y; r[6] = b.z; r[7] = b.w;
  return r;
}
__device__ __forceinline__ v8f wmma_bf16(v8u a, v8u b, v8f c) {
  return __builtin_amdgcn_wmma_f32_16x16x32_bf16(
      false, __builtin_bit_cast(v16bf, a),
      false, __builtin_bit_cast(v16bf, b),
      (short)0, c, false, false);
}

// ---------------------------------------------------------------------------
// Pre-swizzle Vt [32,64,64] (+W [64,32]) into per-lane B-fragment order:
// block (s,t) is 1KB; lane l owns 32B: VGPR r = bf16 pair for
// K = 32s + 16*(l>=16) + 2r, column h = 16t + (l&15).
// ---------------------------------------------------------------------------
__global__ __launch_bounds__(256) void rntn_prep(const float* __restrict__ Vt,
                                                 const float* __restrict__ W,
                                                 unsigned int* __restrict__ vt2) {
  int gid = blockIdx.x * 256 + threadIdx.x;
  if (gid >= NSLAB * 2 * 32 * 8) return;
  int r = gid & 7;
  int l = (gid >> 3) & 31;
  int t = (gid >> 8) & 1;
  int s = gid >> 9;
  int h  = t * 16 + (l & 15);
  int kl = ((l >> 4) << 4) + 2 * r;          // local K pair base 0..30
  float v0, v1;
  if (s < 128) {
    int K = s * 32 + kl;                     // global K = c*64 + d
    int c = K >> 6, d = K & 63;              // pair stays within one c row
    v0 = Vt[((h * 64) + c) * 64 + d];
    v1 = Vt[((h * 64) + c) * 64 + d + 1];
  } else {                                   // linear term: B rows = W[k, h]
    int k = (s - 128) * 32 + kl;
    v0 = W[k * 32 + h];
    v1 = W[(k + 1) * 32 + h];
  }
  vt2[gid] = pk2(v0, v1);
}

// ---------------------------------------------------------------------------
// Leaves: embedding gather (float4 per thread).
// ---------------------------------------------------------------------------
__global__ __launch_bounds__(256) void rntn_embed(const int* __restrict__ ids,
                                                  const float* __restrict__ emb,
                                                  float* __restrict__ leaves) {
  size_t tg = (size_t)blockIdx.x * 256 + threadIdx.x;
  size_t n  = tg >> 3;
  int    c  = (int)(tg & 7);
  int   wid = ids[n];
  ((float4*)leaves)[n * 8 + c] = ((const float4*)emb)[(size_t)wid * 8 + c];
}

// ---------------------------------------------------------------------------
// One tree level: out[g] = tanh(cVc + cW + b), c = in[2g..2g+1] (64 floats).
// Block = 4 waves x 64 nodes; wave = 4 M-tiles x 2 N-tiles of f32 accum.
// ---------------------------------------------------------------------------
__global__ __launch_bounds__(128, 1) void rntn_level(
    const float* __restrict__ in, float* __restrict__ out,
    const unsigned int* __restrict__ vt2, const float* __restrict__ bias) {
  __shared__ unsigned short Cl[4][64 * 64];   // bf16 children, 32KB
  const int w    = threadIdx.x >> 5;
  const int lane = threadIdx.x & 31;
  const size_t g0 = (size_t)blockIdx.x * 256 + (size_t)w * 64;

  // Stage 64 nodes x 64 f32 (contiguous: children of g are rows 2g,2g+1)
  const float4* src = (const float4*)(in + g0 * 64);
  #pragma unroll
  for (int i = 0; i < 32; ++i) {
    int f = i * 32 + lane;
    float4 v = src[f];
    *(uint2*)&Cl[w][f * 4] = make_uint2(pk2(v.x, v.y), pk2(v.z, v.w));
  }
  __syncthreads();

  const int halfsel = lane >> 4;     // A-frag K-half select
  const int rowlane = lane & 15;     // A row / D column lane

  // Plain-C A-fragments: cf[mt][0] covers d 0..31, cf[mt][1] covers d 32..63.
  // K order used here (contiguous, same as vt2 B): lane<16 holds K {0..15}, lane>=16 holds {16..31}.
  v8u cf[4][2];
  #pragma unroll
  for (int mt = 0; mt < 4; ++mt) {
    int row = mt * 16 + rowlane;
    #pragma unroll
    for (int fs = 0; fs < 2; ++fs) {
      int d0 = fs * 32 + halfsel * 16;
      cf[mt][fs] = frag_from(*(const uint4*)&Cl[w][row * 64 + d0],
                             *(const uint4*)&Cl[w][row * 64 + d0 + 8]);
    }
  }

  v8f acc[4][2] = {};
  const uint4* bb = (const uint4*)vt2;

  // 128 bilinear slabs: G[n,K] = C[n, s>>1] * C[n, d], d-half = s&1
  #pragma unroll 2
  for (int s = 0; s < 128; ++s) {
    const uint4* bp = bb + (size_t)s * 128 + lane * 2;   // 2KB per slab
    v8u b0 = frag_from(bp[0],  bp[1]);                   // h 0..15
    v8u b1 = frag_from(bp[64], bp[65]);                  // h 16..31
    __builtin_prefetch(bp + 128, 0, 0);
    const int cs = s >> 1, fs = s & 1;
    #pragma unroll
    for (int mt = 0; mt < 4; ++mt) {
      int row = mt * 16 + rowlane;
      unsigned int sv = Cl[w][row * 64 + cs];
      sv |= sv << 16;
      v8u g;
      #pragma unroll
      for (int r = 0; r < 8; ++r) g[r] = pkmul_bf16(cf[mt][fs][r], sv);
      acc[mt][0] = wmma_bf16(g, b0, acc[mt][0]);
      acc[mt][1] = wmma_bf16(g, b1, acc[mt][1]);
    }
  }

  // 2 linear slabs: A = raw C fragment, B = W rows
  #pragma unroll
  for (int fs = 0; fs < 2; ++fs) {
    const uint4* bp = bb + (size_t)(128 + fs) * 128 + lane * 2;
    v8u b0 = frag_from(bp[0],  bp[1]);
    v8u b1 = frag_from(bp[64], bp[65]);
    #pragma unroll
    for (int mt = 0; mt < 4; ++mt) {
      acc[mt][0] = wmma_bf16(cf[mt][fs], b0, acc[mt][0]);
      acc[mt][1] = wmma_bf16(cf[mt][fs], b1, acc[mt][1]);
    }
  }

  // Epilogue: + b, tanh, store f32 rows (D layout: VGPR r -> M=r(+8 hi lanes))
  #pragma unroll
  for (int t = 0; t < 2; ++t) {
    float bv = bias[t * 16 + rowlane];
    #pragma unroll
    for (int mt = 0; mt < 4; ++mt) {
      #pragma unroll
      for (int r = 0; r < 8; ++r) {
        size_t nrow = g0 + (size_t)(mt * 16 + r + halfsel * 8);
        out[nrow * 32 + t * 16 + rowlane] = tanhf(acc[mt][t][r] + bv);
      }
    }
  }
}

// ---------------------------------------------------------------------------
// Classifier: logits = nodes @ W_out + b_out, then log_softmax. 1 row/thread.
// ---------------------------------------------------------------------------
__global__ __launch_bounds__(256) void rntn_out(const float* __restrict__ nodes,
                                                const float* __restrict__ Wout,
                                                const float* __restrict__ bout,
                                                float* __restrict__ outp) {
  __shared__ float sW[Ht * OUTC];
  __shared__ float sb[OUTC];
  if (threadIdx.x < Ht * OUTC) sW[threadIdx.x] = Wout[threadIdx.x];
  if (threadIdx.x < OUTC)      sb[threadIdx.x] = bout[threadIdx.x];
  __syncthreads();

  size_t row = (size_t)blockIdx.x * 256 + threadIdx.x;
  const float4* h4 = (const float4*)(nodes + row * Ht);
  float lg[OUTC];
  #pragma unroll
  for (int o = 0; o < OUTC; ++o) lg[o] = sb[o];
  #pragma unroll
  for (int q = 0; q < 8; ++q) {
    float4 hv = h4[q];
    const float* wr = &sW[q * 4 * OUTC];
    #pragma unroll
    for (int o = 0; o < OUTC; ++o)
      lg[o] += hv.x * wr[o] + hv.y * wr[OUTC + o] +
               hv.z * wr[2 * OUTC + o] + hv.w * wr[3 * OUTC + o];
  }
  float m = lg[0];
  #pragma unroll
  for (int o = 1; o < OUTC; ++o) m = fmaxf(m, lg[o]);
  float se = 0.f;
  #pragma unroll
  for (int o = 0; o < OUTC; ++o) se += __expf(lg[o] - m);
  float ls = m + __logf(se);
  #pragma unroll
  for (int o = 0; o < OUTC; ++o) outp[row * OUTC + o] = lg[o] - ls;
}

// ---------------------------------------------------------------------------
extern "C" void kernel_launch(void* const* d_in, const int* in_sizes, int n_in,
                              void* d_out, int out_size, void* d_ws, size_t ws_size,
                              hipStream_t stream) {
  const int*   word_ids = (const int*)d_in[0];
  const float* embed    = (const float*)d_in[1];
  const float* Vt       = (const float*)d_in[2];
  const float* W        = (const float*)d_in[3];
  const float* b        = (const float*)d_in[4];
  const float* Wout     = (const float*)d_in[5];
  const float* bout     = (const float*)d_in[6];
  float*       outp     = (float*)d_out;

  // Workspace: node buffer [B*127, 32] f32 (levels concatenated bottom-up),
  // then swizzled bf16 Vt2 B-fragments (260KB).
  float* nodes = (float*)d_ws;
  size_t nodesBytes = (size_t)Bt * 127 * Ht * sizeof(float);
  unsigned int* vt2 =
      (unsigned int*)((char*)d_ws + ((nodesBytes + 255) & ~(size_t)255));

  rntn_prep<<<(NSLAB * 2 * 32 * 8 + 255) / 256, 256, 0, stream>>>(Vt, W, vt2);
  rntn_embed<<<(Bt * Lt * 8) / 256, 256, 0, stream>>>(word_ids, embed, nodes);

  size_t inOff = 0, rowsIn = (size_t)Bt * Lt;
  for (int l = 0; l < 6; ++l) {
    size_t rowsOut = rowsIn >> 1;
    size_t outOff  = inOff + rowsIn;
    rntn_level<<<(unsigned)(rowsOut / 256), 128, 0, stream>>>(
        nodes + inOff * Ht, nodes + outOff * Ht, vt2, b);
    inOff  = outOff;
    rowsIn = rowsOut;
  }
  rntn_out<<<(unsigned)(((size_t)Bt * 127) / 256), 256, 0, stream>>>(
      nodes, Wout, bout, outp);
}
